// LSTMTagger_9105330667784
// MI455X (gfx1250) — compile-verified, matrix operand order checked by reading, not run on hardware
//
#include <hip/hip_runtime.h>
#include <hip/hip_bf16.h>

// ---------------------------------------------------------------------------
// LSTM tagger for MI455X (gfx1250, wave32).
//   T=2048 steps, D=1024 in, H=2048 hidden, TAGS=1024.
// Phase 1: x_gates = x @ W_ih^T + (b_ih+b_hh)  -> WMMA bf16 GEMM, B tiles
//          staged into LDS with the Tensor Data Mover (double-buffered).
// Phase 2: sequential LSTM; W_hh (32MB bf16, pre-swizzled to fragment order
//          in global) pinned in LDS across 128 persistent WGs (256KB/WG,
//          loaded with one 1-D TDM descriptor each); per-step GEMV done with
//          WMMA via broadcast-B; grid-wide atomic barrier between steps.
// Phase 3: tag = hs @ W_out^T + b_out (WMMA + TDM staging); log_softmax.
// ---------------------------------------------------------------------------

typedef unsigned short u16;
typedef __attribute__((ext_vector_type(16))) __bf16 v16bf;
typedef __attribute__((ext_vector_type(8)))  float  v8f;
typedef __attribute__((ext_vector_type(4))) unsigned int v4u;
typedef __attribute__((ext_vector_type(8))) int v8i_;
typedef __attribute__((ext_vector_type(4))) int v4i_;

union FragB16 { uint4 u[2]; v16bf v; };

#define Tn    2048
#define Dn    1024
#define Hn    2048
#define G4n   8192     // 4*H
#define TAGSn 1024
#define NWG   128      // persistent workgroups for recurrence (H/16)

__device__ __forceinline__ u16 f2bf(float x) {
  unsigned u = __float_as_uint(x);
  unsigned r = (u + 0x7FFFu + ((u >> 16) & 1u)) >> 16;   // RNE
  return (u16)r;
}

// Load a 16x32 bf16 fragment (A layout; B mirrors it) from a row-major
// [rows x ld] bf16 matrix (global or LDS).  lane&15 = row, lane>>4 = K-half.
__device__ __forceinline__ FragB16 load_frag(const u16* base, int ld,
                                             int row0, int k0, int lane) {
  int r = row0 + (lane & 15);
  int half = lane >> 4;
  const u16* p = base + (size_t)r * ld + k0 + 8 * half;
  FragB16 f;
  f.u[0] = *(const uint4*)(p);        // K = k0+8h .. +7
  f.u[1] = *(const uint4*)(p + 16);   // K = k0+16+8h .. +7
  return f;
}

__device__ __forceinline__ v8f wmma_bf16(const FragB16& a, const FragB16& b,
                                         v8f c) {
  return __builtin_amdgcn_wmma_f32_16x16x32_bf16(
      false, a.v, false, b.v, (short)0, c, false, false);
}

// --------------------------------------------------------------- TDM issue
// 2-D tile DMA: global (row-major, stride0 in 8B units) -> LDS (compact).
// D# built per CDNA5 ISA (group0: count/lds/global/type,
// group1: data_size=8B, dims, tile dims, stride; groups 2/3 zero => <=2D).
__device__ __forceinline__ void tdm_load_2d(unsigned long long gaddr,
                                            unsigned lds_off,
                                            unsigned td0, unsigned td1,
                                            unsigned tile0, unsigned tile1,
                                            unsigned stride0) {
  v4u g0;
  g0[0] = 1u;                                   // count=1, user mode
  g0[1] = lds_off;                              // lds_addr
  g0[2] = (unsigned)gaddr;                      // global_addr[31:0]
  g0[3] = ((unsigned)(gaddr >> 32) & 0x01FFFFFFu) | (2u << 30); // [56:32]|type=2
  v8i_ g1;
  g1[0] = (int)(3u << 16);                      // data_size = 8 bytes
  g1[1] = (int)((td0 & 0xFFFFu) << 16);         // tensor_dim0[15:0]
  g1[2] = (int)((td0 >> 16) | ((td1 & 0xFFFFu) << 16));
  g1[3] = (int)((td1 >> 16) | (tile0 << 16));   // tile_dim0
  g1[4] = (int)tile1;                           // tile_dim1
  g1[5] = (int)stride0;                         // tensor_dim0_stride[31:0]
  g1[6] = 0;
  g1[7] = 0;
  v4i_ g2 = {0, 0, 0, 0}, g3 = {0, 0, 0, 0};
#if defined(__clang_major__) && (__clang_major__ >= 23)
  v8i_ g4 = {0, 0, 0, 0, 0, 0, 0, 0};
  __builtin_amdgcn_tensor_load_to_lds(g0, g1, g2, g3, g4, 0);
#else
  __builtin_amdgcn_tensor_load_to_lds(g0, g1, g2, g3, 0);
#endif
}

// ---------------------------------------------------------------- fp32->bf16
__global__ void cvt_bf16_kernel(const float* __restrict__ src,
                                u16* __restrict__ dst, int n) {
  for (int i = blockIdx.x * blockDim.x + threadIdx.x; i < n;
       i += gridDim.x * blockDim.x)
    dst[i] = f2bf(src[i]);
}

// ---------------------------------------------- W_hh: fp32 -> swizzled bf16
// Writes W_hh in per-WG fragment order: WG g owns hidden units [g*16,g*16+16);
// slice = 4 gate tiles x 64 K-chunks x (32 lanes x 32B) = 256KB, contiguous.
__global__ void swizzle_whh_kernel(const float* __restrict__ whh,
                                   char* __restrict__ dst) {
  int id = blockIdx.x * blockDim.x + threadIdx.x;
  if (id >= NWG * 16384) return;
  int g = id >> 14;
  int c = id & 16383;              // (t4,kc,lane,part) -> one 16B chunk
  int part = c & 1, ln = (c >> 1) & 31, kc = (c >> 6) & 63, t4 = c >> 12;
  int row = t4 * Hn + g * 16 + (ln & 15);
  int k   = kc * 32 + part * 16 + 8 * (ln >> 4);
  const float* s = whh + (size_t)row * Hn + k;
  union { u16 h[8]; uint4 v; } tmp;
#pragma unroll
  for (int i = 0; i < 8; ++i) tmp.h[i] = f2bf(s[i]);
  *(uint4*)(dst + (size_t)g * 262144 +
            (size_t)(t4 * 64 + kc) * 1024 + ln * 32 + part * 16) = tmp.v;
}

// ------------------------------------------------- Phase 1: x_gates (GEMM)
// grid = (8192/64, 2048/128), block = 256 (8 waves). Wave computes 16Mx64N.
// B tile (64 rows x 32 K bf16 = 4KB) staged in LDS by TDM, double-buffered.
__global__ __launch_bounds__(256)
void gemm_xgates_kernel(const u16* __restrict__ xb,     // [2048][1024]
                        const u16* __restrict__ wihb,   // [8192][1024]
                        const float* __restrict__ bih,
                        const float* __restrict__ bhh,
                        float* __restrict__ xg) {       // [2048][8192]
  extern __shared__ char sm[];                          // 2 x 4KB B buffers
  const int wv = threadIdx.x >> 5, lane = threadIdx.x & 31, half = lane >> 4;
  const int m0 = blockIdx.y * 128 + wv * 16;
  const int n0 = blockIdx.x * 64;
  const int KITER = Dn / 32;

  unsigned long long wbase = (unsigned long long)(uintptr_t)wihb;
  if (wv == 0)   // prologue: stage B tile for k=0 into buffer 0
    tdm_load_2d(wbase + ((size_t)n0 * Dn) * 2, 0u,
                /*td0=*/Dn / 4, /*td1=*/G4n, /*tile0=*/8, /*tile1=*/64,
                /*stride0=*/Dn / 4);

  v8f acc0 = {}, acc1 = {}, acc2 = {}, acc3 = {};
  FragB16 a = load_frag(xb, Dn, m0, 0, lane);
  for (int kk = 0; kk < KITER; ++kk) {
    if (wv == 0) {
      if (kk + 1 < KITER) {
        tdm_load_2d(wbase + ((size_t)n0 * Dn + (kk + 1) * 32) * 2,
                    (unsigned)(((kk + 1) & 1) * 4096),
                    Dn / 4, G4n, 8, 64, Dn / 4);
        __builtin_amdgcn_s_wait_tensorcnt(1);   // current tile done
      } else {
        __builtin_amdgcn_s_wait_tensorcnt(0);
      }
    }
    __syncthreads();
    const u16* bt = (const u16*)(sm + (kk & 1) * 4096);   // 64 x 32 bf16
    FragB16 an;
    if (kk + 1 < KITER) an = load_frag(xb, Dn, m0, (kk + 1) * 32, lane);
    FragB16 b0 = load_frag(bt, 32,  0, 0, lane);
    FragB16 b1 = load_frag(bt, 32, 16, 0, lane);
    FragB16 b2 = load_frag(bt, 32, 32, 0, lane);
    FragB16 b3 = load_frag(bt, 32, 48, 0, lane);
    acc0 = wmma_bf16(a, b0, acc0);
    acc1 = wmma_bf16(a, b1, acc1);
    acc2 = wmma_bf16(a, b2, acc2);
    acc3 = wmma_bf16(a, b3, acc3);
    a = an;
    __syncthreads();     // all waves done with buf[kk&1] before next DMA
  }
  v8f accs[4] = {acc0, acc1, acc2, acc3};
#pragma unroll
  for (int j = 0; j < 4; ++j) {
    int ncol = n0 + j * 16 + (lane & 15);
    float bias = bih[ncol] + bhh[ncol];
#pragma unroll
    for (int r = 0; r < 8; ++r)
      xg[(size_t)(m0 + r + 8 * half) * G4n + ncol] = accs[j][r] + bias;
  }
}

// ------------------------------------------------ barrier counter reset
__global__ void zero_bar_kernel(unsigned* bar) {
  if (threadIdx.x == 0 && blockIdx.x == 0) *bar = 0u;
}

// ------------------------------------------------- Phase 2: recurrence
// 128 persistent WGs x 256 threads.  WG g owns hidden units [g*16, g*16+16).
// LDS: 256KB swizzled W_hh slice (one 1-D TDM copy) + 4KB h stage + scratch.
#define HOFF (4 * 64 * 1024)            // 262144: staged h (2048 bf16)
#define ROFF (HOFF + 4096)              // 64 f32 gate partials
#define COFF (ROFF + 256)               // 16 f32 cell state
#define SMEM_BYTES (COFF + 64)

__global__ __launch_bounds__(256)
void lstm_kernel(const float* __restrict__ xg,     // [2048][8192]
                 const char* __restrict__ whhswz,  // [128][262144] swizzled
                 u16* __restrict__ hcur,           // [2048]
                 u16* __restrict__ hsb,            // [2048][2048]
                 unsigned* __restrict__ bar) {
  extern __shared__ char sm[];
  u16*   hlds = (u16*)(sm + HOFF);
  float* red  = (float*)(sm + ROFF);
  float* cst  = (float*)(sm + COFF);

  const int tid  = threadIdx.x;
  const int wv   = tid >> 5, lane = tid & 31, half = lane >> 4;
  const int j0   = blockIdx.x * 16;

  // --- TDM: pull this WG's pre-swizzled 256KB W_hh slice into LDS.
  //     1-D tensor, data_size=8B, 32768 elements.
  if (wv == 0) {
    unsigned long long ga = (unsigned long long)(uintptr_t)
                            (whhswz + (size_t)blockIdx.x * 262144);
    tdm_load_2d(ga, 0u, /*td0=*/32768u, /*td1=*/1u,
                /*tile0=*/32768u, /*tile1=*/1u, /*stride0=*/32768u);
    __builtin_amdgcn_s_wait_tensorcnt(0);
  }
  if (tid < 16) cst[tid] = 0.0f;
  __syncthreads();

  for (int t = 0; t < Tn; ++t) {
    // stage h[t-1] (zeros at t==0): 2048 bf16, one uint4 per thread
    uint4 hv;
    if (t == 0) { hv.x = hv.y = hv.z = hv.w = 0u; }
    else        { hv = *(const uint4*)(hcur + tid * 8); }
    *(uint4*)(hlds + tid * 8) = hv;
    if (tid < 64) red[tid] = 0.0f;
    __syncthreads();

    // GEMV via WMMA: wave wv handles K-chunks [wv*8, wv*8+8) for all 4 gates.
    v8f a0 = {}, a1 = {}, a2 = {}, a3 = {};
#pragma unroll 2
    for (int kc8 = 0; kc8 < 8; ++kc8) {
      int kc = wv * 8 + kc8;
      // broadcast-B: B[k][n] = h[k] for all n (lane-uniform LDS reads)
      FragB16 b;
      const u16* hp = hlds + kc * 32 + 8 * half;
      b.u[0] = *(const uint4*)(hp);
      b.u[1] = *(const uint4*)(hp + 16);
      const char* ap = sm + (size_t)(0 * 64 + kc) * 1024 + lane * 32;
      FragB16 a;
      a.u[0] = *(const uint4*)(ap);       a.u[1] = *(const uint4*)(ap + 16);
      a0 = wmma_bf16(a, b, a0);
      ap = sm + (size_t)(1 * 64 + kc) * 1024 + lane * 32;
      a.u[0] = *(const uint4*)(ap);       a.u[1] = *(const uint4*)(ap + 16);
      a1 = wmma_bf16(a, b, a1);
      ap = sm + (size_t)(2 * 64 + kc) * 1024 + lane * 32;
      a.u[0] = *(const uint4*)(ap);       a.u[1] = *(const uint4*)(ap + 16);
      a2 = wmma_bf16(a, b, a2);
      ap = sm + (size_t)(3 * 64 + kc) * 1024 + lane * 32;
      a.u[0] = *(const uint4*)(ap);       a.u[1] = *(const uint4*)(ap + 16);
      a3 = wmma_bf16(a, b, a3);
    }
    // every column of D holds the GEMV result: reduce column 0 across waves
    if ((lane & 15) == 0) {
#pragma unroll
      for (int r = 0; r < 8; ++r) {
        atomicAdd(&red[ 0 + r + 8 * half], a0[r]);
        atomicAdd(&red[16 + r + 8 * half], a1[r]);
        atomicAdd(&red[32 + r + 8 * half], a2[r]);
        atomicAdd(&red[48 + r + 8 * half], a3[r]);
      }
    }
    __syncthreads();

    // pointwise cell update for this WG's 16 hidden units
    if (tid < 16) {
      int j = j0 + tid;
      const float* xr = xg + (size_t)t * G4n;
      float gi = xr[         j] + red[     tid];
      float gf = xr[  Hn   + j] + red[16 + tid];
      float gg = xr[2*Hn   + j] + red[32 + tid];
      float go = xr[3*Hn   + j] + red[48 + tid];
      gi = 1.0f / (1.0f + __expf(-gi));
      gf = 1.0f / (1.0f + __expf(-gf));
      go = 1.0f / (1.0f + __expf(-go));
      gg = tanhf(gg);
      float c = gf * cst[tid] + gi * gg;
      cst[tid] = c;
      float h = go * tanhf(c);
      u16 hb = f2bf(h);
      hcur[j] = hb;
      hsb[(size_t)t * Hn + j] = hb;
      __threadfence();                 // device-scope release of h
    }
    __syncthreads();

    // grid-wide step barrier (persistent kernel; monotonic counter)
    if (tid == 0) {
      atomicAdd(bar, 1u);
      unsigned target = (unsigned)(t + 1) * NWG;
      while (__hip_atomic_load(bar, __ATOMIC_ACQUIRE,
                               __HIP_MEMORY_SCOPE_AGENT) < target)
        __builtin_amdgcn_s_sleep(1);
      __threadfence();                 // acquire all WGs' h stores
    }
    __syncthreads();
  }
}

// ------------------------------------------------- Phase 3: output GEMM
// grid = (1024/64, 2048/128), block = 256.  Same TDM-staged-B structure.
__global__ __launch_bounds__(256)
void gemm_out_kernel(const u16* __restrict__ hsb,    // [2048][2048]
                     const u16* __restrict__ woutb,  // [1024][2048]
                     const float* __restrict__ bout,
                     float* __restrict__ out) {      // [2048][1024]
  extern __shared__ char sm[];
  const int wv = threadIdx.x >> 5, lane = threadIdx.x & 31, half = lane >> 4;
  const int m0 = blockIdx.y * 128 + wv * 16;
  const int n0 = blockIdx.x * 64;
  const int KITER = Hn / 32;

  unsigned long long wbase = (unsigned long long)(uintptr_t)woutb;
  if (wv == 0)
    tdm_load_2d(wbase + ((size_t)n0 * Hn) * 2, 0u,
                Hn / 4, TAGSn, 8, 64, Hn / 4);

  v8f acc0 = {}, acc1 = {}, acc2 = {}, acc3 = {};
  FragB16 a = load_frag(hsb, Hn, m0, 0, lane);
  for (int kk = 0; kk < KITER; ++kk) {
    if (wv == 0) {
      if (kk + 1 < KITER) {
        tdm_load_2d(wbase + ((size_t)n0 * Hn + (kk + 1) * 32) * 2,
                    (unsigned)(((kk + 1) & 1) * 4096),
                    Hn / 4, TAGSn, 8, 64, Hn / 4);
        __builtin_amdgcn_s_wait_tensorcnt(1);
      } else {
        __builtin_amdgcn_s_wait_tensorcnt(0);
      }
    }
    __syncthreads();
    const u16* bt = (const u16*)(sm + (kk & 1) * 4096);
    FragB16 an;
    if (kk + 1 < KITER) an = load_frag(hsb, Hn, m0, (kk + 1) * 32, lane);
    FragB16 b0 = load_frag(bt, 32,  0, 0, lane);
    FragB16 b1 = load_frag(bt, 32, 16, 0, lane);
    FragB16 b2 = load_frag(bt, 32, 32, 0, lane);
    FragB16 b3 = load_frag(bt, 32, 48, 0, lane);
    acc0 = wmma_bf16(a, b0, acc0);
    acc1 = wmma_bf16(a, b1, acc1);
    acc2 = wmma_bf16(a, b2, acc2);
    acc3 = wmma_bf16(a, b3, acc3);
    a = an;
    __syncthreads();
  }
  v8f accs[4] = {acc0, acc1, acc2, acc3};
#pragma unroll
  for (int j = 0; j < 4; ++j) {
    int ncol = n0 + j * 16 + (lane & 15);
    float bias = bout[ncol];
#pragma unroll
    for (int r = 0; r < 8; ++r)
      out[(size_t)(m0 + r + 8 * half) * TAGSn + ncol] = accs[j][r] + bias;
  }
}

// ------------------------------------------------- log_softmax (in place)
__global__ __launch_bounds__(256)
void logsoftmax_kernel(float* __restrict__ out) {
  __shared__ float s[256];
  int tid = threadIdx.x;
  float* p = out + (size_t)blockIdx.x * TAGSn;
  float m = -3.402823466e38f;
  for (int i = tid; i < TAGSn; i += 256) m = fmaxf(m, p[i]);
  s[tid] = m; __syncthreads();
  for (int st = 128; st > 0; st >>= 1) {
    if (tid < st) s[tid] = fmaxf(s[tid], s[tid + st]);
    __syncthreads();
  }
  float mx = s[0]; __syncthreads();
  float sum = 0.0f;
  for (int i = tid; i < TAGSn; i += 256) sum += __expf(p[i] - mx);
  s[tid] = sum; __syncthreads();
  for (int st = 128; st > 0; st >>= 1) {
    if (tid < st) s[tid] += s[tid + st];
    __syncthreads();
  }
  float lse = logf(s[0]); __syncthreads();
  for (int i = tid; i < TAGSn; i += 256) p[i] = p[i] - mx - lse;
}

// ---------------------------------------------------------------------------
extern "C" void kernel_launch(void* const* d_in, const int* in_sizes, int n_in,
                              void* d_out, int out_size, void* d_ws, size_t ws_size,
                              hipStream_t stream) {
  const float* x    = (const float*)d_in[0];   // [2048,1024]
  const float* Wih  = (const float*)d_in[1];   // [8192,1024]
  const float* Whh  = (const float*)d_in[2];   // [8192,2048]
  const float* bih  = (const float*)d_in[3];   // [8192]
  const float* bhh  = (const float*)d_in[4];   // [8192]
  const float* Wout = (const float*)d_in[5];   // [1024,2048]
  const float* bout = (const float*)d_in[6];   // [1024]
  float* out = (float*)d_out;                  // [2048,1024]

  char* ws = (char*)d_ws;
  size_t off = 0;
  u16* xb      = (u16*)(ws + off); off += (size_t)Tn * Dn * 2;      // 4 MB
  u16* wihb    = (u16*)(ws + off); off += (size_t)G4n * Dn * 2;     // 16 MB
  char* whhswz = (char*)(ws + off); off += (size_t)NWG * 262144;    // 32 MB
  u16* woutb   = (u16*)(ws + off); off += (size_t)TAGSn * Hn * 2;   // 4 MB
  float* xg    = (float*)(ws + off); off += (size_t)Tn * G4n * 4;   // 64 MB
  u16* hsb     = (u16*)(ws + off); off += (size_t)Tn * Hn * 2;      // 8 MB
  u16* hcur    = (u16*)(ws + off); off += 4096;
  unsigned* bar = (unsigned*)(ws + off); off += 256;
  (void)ws_size; (void)in_sizes; (void)n_in; (void)out_size;

  // fp32 -> bf16 conversions (+ fused swizzle for W_hh)
  cvt_bf16_kernel<<<4096, 256, 0, stream>>>(x,    xb,    Tn * Dn);
  cvt_bf16_kernel<<<4096, 256, 0, stream>>>(Wih,  wihb,  G4n * Dn);
  cvt_bf16_kernel<<<4096, 256, 0, stream>>>(Wout, woutb, TAGSn * Hn);
  swizzle_whh_kernel<<<(NWG * 16384) / 256, 256, 0, stream>>>(Whh, whhswz);

  // Phase 1: x_gates
  gemm_xgates_kernel<<<dim3(G4n / 64, Tn / 128), 256, 8192, stream>>>(
      xb, wihb, bih, bhh, xg);

  // Phase 2: recurrence (persistent, LDS-resident W_hh via TDM)
  zero_bar_kernel<<<1, 32, 0, stream>>>(bar);
  lstm_kernel<<<NWG, 256, SMEM_BYTES, stream>>>(xg, whhswz, hcur, hsb, bar);

  // Phase 3: projection + log_softmax
  gemm_out_kernel<<<dim3(TAGSn / 64, Tn / 128), 256, 8192, stream>>>(
      hsb, woutb, bout, out);
  logsoftmax_kernel<<<Tn, 256, 0, stream>>>(out);
}
